// HMM_48223892799789
// MI455X (gfx1250) — matrix operand order repaired and checked by reading, not compiled, not run on hardware
//
#include <hip/hip_runtime.h>
#include <hip/hip_bf16.h>

typedef __attribute__((ext_vector_type(16))) _Float16 v16h;
typedef __attribute__((ext_vector_type(8)))  float    v8f;
typedef __attribute__((ext_vector_type(4)))  float    v4f;

#define VOCAB 32000
#define NS    64
#define BATCH 256
#define TLEN  1024

#define LOG2E 1.4426950408889634f
#define LN2   0.6931471805599453f

// d_ws float-offset layout
#define WS_LOGZ 0                 // 64 floats: logsumexp over vocab per state
#define WS_PB0  64                // 64 floats: softmax(begin)  (linear!)
#define WS_TT   128               // 2048 floats of space = 4096 f16: Texp^T [j][i]
#define WS_PM   2176              // 125*64 partial maxima
#define WS_PS   (2176 + 8000)     // 125*64 partial sums
#define WS_EXP  18432             // optional: VOCAB*NS floats of exp(emb - logZ)

// ---------------------------------------------------------------------------
// Cross-half (lane i <-> lane i^16) exchange. Prefer v_permlanex16_b32 (VALU)
// over the ds_bpermute that __shfl_xor lowers to.
// ---------------------------------------------------------------------------
__device__ __forceinline__ float other_half(float v) {
#if __has_builtin(__builtin_amdgcn_permlanex16)
    return __int_as_float(__builtin_amdgcn_permlanex16(
        __float_as_int(v), __float_as_int(v),
        0x76543210, 0xfedcba98, false, false));   // identity lane map
#else
    return __shfl_xor(v, 16, 32);
#endif
}

// ---------------------------------------------------------------------------
// 16-lane (half-wave) all-reduce sum via DPP row rotates: pure VALU, no DS.
// ---------------------------------------------------------------------------
__device__ __forceinline__ float halfrow_sum(float S) {
    S += __int_as_float(__builtin_amdgcn_update_dpp(
            0, __float_as_int(S), 0x121, 0xf, 0xf, true));   // row_ror:1
    S += __int_as_float(__builtin_amdgcn_update_dpp(
            0, __float_as_int(S), 0x122, 0xf, 0xf, true));   // row_ror:2
    S += __int_as_float(__builtin_amdgcn_update_dpp(
            0, __float_as_int(S), 0x124, 0xf, 0xf, true));   // row_ror:4
    S += __int_as_float(__builtin_amdgcn_update_dpp(
            0, __float_as_int(S), 0x128, 0xf, 0xf, true));   // row_ror:8
    return S;
}

// ---------------------------------------------------------------------------
// Kernel 1a: per-block online logsumexp partials over vocab axis.
// ---------------------------------------------------------------------------
__global__ void __launch_bounds__(256) k_logz_part(const float* __restrict__ emb,
                                                   float* __restrict__ pm,
                                                   float* __restrict__ ps) {
    const int tid = threadIdx.x;
    const int s   = tid & 63;
    const int vg  = tid >> 6;
    const int vb  = blockIdx.x * 256;
    float m = -INFINITY, sum = 0.f;
#pragma unroll 4
    for (int i = 0; i < 64; ++i) {
        const int v = vb + i * 4 + vg;
        const float x = emb[v * NS + s];
        if (x > m) { sum = sum * __expf(m - x) + 1.f; m = x; }
        else       { sum += __expf(x - m); }
    }
    __shared__ float sm[256], ss[256];
    sm[tid] = m; ss[tid] = sum;
    __syncthreads();
    if (tid < 128) {
        float m2 = sm[tid + 128], s2 = ss[tid + 128];
        float M = fmaxf(sm[tid], m2);
        ss[tid] = ss[tid] * __expf(sm[tid] - M) + s2 * __expf(m2 - M);
        sm[tid] = M;
    }
    __syncthreads();
    if (tid < 64) {
        float m2 = sm[tid + 64], s2 = ss[tid + 64];
        float M = fmaxf(sm[tid], m2);
        float S = ss[tid] * __expf(sm[tid] - M) + s2 * __expf(m2 - M);
        pm[blockIdx.x * 64 + tid] = M;
        ps[blockIdx.x * 64 + tid] = S;
    }
}

// ---------------------------------------------------------------------------
// Kernel 1b: combine 125 partials per state -> logZ[s]
// ---------------------------------------------------------------------------
__global__ void __launch_bounds__(64) k_logz_final(const float* __restrict__ pm,
                                                   const float* __restrict__ ps,
                                                   float* __restrict__ logZ) {
    const int s = threadIdx.x;
    float m = -INFINITY, sum = 0.f;
    for (int b = 0; b < 125; ++b) {
        const float m2 = pm[b * 64 + s], s2 = ps[b * 64 + s];
        const float M = fmaxf(m, m2);
        sum = sum * __expf(m - M) + s2 * __expf(m2 - M);
        m = M;
    }
    logZ[s] = m + __logf(sum);
}

// ---------------------------------------------------------------------------
// Kernel 2: Texp^T (f16, [j][i]) = softmax(transition, -1)^T; softmax(begin)
// (linear); zero d_out.
// ---------------------------------------------------------------------------
__global__ void __launch_bounds__(64) k_prep(const float* __restrict__ trans,
                                             const float* __restrict__ begin,
                                             float* __restrict__ wsf,
                                             float* __restrict__ out) {
    const int i = threadIdx.x;
    const float* row = trans + i * NS;
    float m = -INFINITY;
    for (int j = 0; j < NS; ++j) m = fmaxf(m, row[j]);
    float s = 0.f;
    for (int j = 0; j < NS; ++j) s += __expf(row[j] - m);
    const float inv = 1.f / s;
    _Float16* Tt = (_Float16*)(wsf + WS_TT);
    for (int j = 0; j < NS; ++j)
        Tt[j * NS + i] = (_Float16)(__expf(row[j] - m) * inv);

    __shared__ float sb[64];
    __shared__ float red[2];
    sb[i] = begin[i];
    __syncthreads();
    if (i == 0) {
        float M = -INFINITY;
        for (int k = 0; k < NS; ++k) M = fmaxf(M, sb[k]);
        float S = 0.f;
        for (int k = 0; k < NS; ++k) S += __expf(sb[k] - M);
        red[0] = M; red[1] = 1.f / S;
        out[0] = 0.f;
    }
    __syncthreads();
    wsf[WS_PB0 + i] = __expf(sb[i] - red[0]) * red[1];   // linear softmax(begin)
}

// ---------------------------------------------------------------------------
// Kernel 2b (optional fast path): expE[v][s] = exp(emb[v][s] - logZ[s])
// ---------------------------------------------------------------------------
__global__ void __launch_bounds__(256) k_expemb(const float* __restrict__ emb,
                                                const float* __restrict__ logZ,
                                                float* __restrict__ expE) {
    const int idx = blockIdx.x * 256 + threadIdx.x;   // VOCAB*NS total
    const int s = idx & 63;
    expE[idx] = __expf(emb[idx] - logZ[s]);
}

// ---------------------------------------------------------------------------
// Main kernel: 16 blocks x 32 threads (1 wave), 16 batch chains per wave.
// Orientation: D = Texp^T @ p^T  (M = state tiles, N = batch columns).
// State relabeling baked into A: tile jt, row m -> state
//   m<8 : {0,8,32,40}[jt] + m      (half-0 lanes)
//   m>=8: {16,24,48,56}[jt] + m-8  (half-1 lanes)
// D's register layout == next step's B operand layout after in-lane f16
// packing: no LDS, no barriers; one cross-half exchange per step for S.
// ---------------------------------------------------------------------------
template <bool PRE>
__global__ void __launch_bounds__(32) k_hmm(const int*   __restrict__ sent,
                                            const float* __restrict__ masks,
                                            const float* __restrict__ emb,
                                            const float* __restrict__ wsf,
                                            float*       __restrict__ out) {
    const int lane = threadIdx.x;
    const int half = lane >> 4;
    const int lm   = lane & 15;
    const int b0   = blockIdx.x * 16;

    const float* ebase = PRE ? (wsf + WS_EXP) : emb;
    const _Float16* Tt = (const _Float16*)(wsf + WS_TT);   // TexpT[j][i], 64x64

    const int base0[4] = {0, 8, 32, 40};    // states of tile jt, rows 0-7
    const int base1[4] = {16, 24, 48, 56};  // states of tile jt, rows 8-15

    // A fragments A[jt][kt] (16x32 f16), resident: A[m][k] = Texp[i=K][j=state].
    v16h Af[4][2];
#pragma unroll
    for (int jt = 0; jt < 4; ++jt) {
        const int jrow = (lm < 8) ? (base0[jt] + lm) : (base1[jt] + lm - 8);
#pragma unroll
        for (int kt = 0; kt < 2; ++kt) {
            const _Float16* p = Tt + jrow * NS + kt * 32 + 16 * half;
            union { v16h v; v4f f[2]; } u;
            u.f[0] = *(const v4f*)(p);
            u.f[1] = *(const v4f*)(p + 8);
            Af[jt][kt] = u.v;
        }
    }

    // This lane's 32 states: st(jt, r) = (half ? base1 : base0)[jt] + r.
    float mid[4][8];
    float lzv[32];                        // logZ*log2e per state (only !PRE)
#pragma unroll
    for (int jt = 0; jt < 4; ++jt) {
        const int sb = half ? base1[jt] : base0[jt];
#pragma unroll
        for (int r = 0; r < 8; ++r) {
            mid[jt][r] = wsf[WS_PB0 + sb + r];
            lzv[jt * 8 + r] = PRE ? 0.f : wsf[WS_LOGZ + sb + r] * LOG2E;
        }
    }

    // Emission gather: lane's states are two contiguous 16-float runs of the
    // token's row: [32g + 16*half .. +15], g = 0,1  ->  8 b128 loads.
    auto loadE = [&](int tokv, float* E) {
        const float* er = ebase + (unsigned)tokv * (unsigned)NS + 16 * half;
#pragma unroll
        for (int g = 0; g < 2; ++g) {
            v4f x[4];
            x[0] = *(const v4f*)(er + 32 * g);
            x[1] = *(const v4f*)(er + 32 * g + 4);
            x[2] = *(const v4f*)(er + 32 * g + 8);
            x[3] = *(const v4f*)(er + 32 * g + 12);
#pragma unroll
            for (int c = 0; c < 4; ++c)
#pragma unroll
                for (int e = 0; e < 4; ++e) {
                    const int idx = (2 * g + (c >> 1)) * 8 + (c & 1) * 4 + e;
                    const float v = x[c][e];
                    E[idx] = PRE ? v
                        : __builtin_amdgcn_exp2f(
                              __builtin_fmaf(v, LOG2E, -lzv[idx]));
                }
        }
    };

    const int rowoff = (b0 + lm) * TLEN;   // this lane's batch row
    float pacc = 0.f;
    float Ecur[32], Enext[32];
    int tok0 = sent[rowoff];
    int tokN = sent[rowoff + 1];
    loadE(tok0, Ecur);

    for (int t = 0; t < TLEN; ++t) {
        // q = mid .* E ; S = full 64-state sum for this batch column.
        float q[4][8];
        float part[4];
#pragma unroll
        for (int jt = 0; jt < 4; ++jt) {
#pragma unroll
            for (int r = 0; r < 8; ++r) q[jt][r] = mid[jt][r] * Ecur[jt * 8 + r];
            part[jt] = ((q[jt][0] + q[jt][1]) + (q[jt][2] + q[jt][3]))
                     + ((q[jt][4] + q[jt][5]) + (q[jt][6] + q[jt][7]));
        }
        const float Sl = (part[0] + part[1]) + (part[2] + part[3]);
        const float S  = Sl + other_half(Sl);   // add other half's 32 states

        pacc = __builtin_fmaf(__builtin_amdgcn_logf(S) * LN2,
                              masks[rowoff + t], pacc);

        // Pipeline: gather E for t+1, refill token for t+2.
        if (t + 1 < TLEN) loadE(tokN, Enext);
        if (t + 2 < TLEN) tokN = sent[rowoff + t + 2];

        if (t < TLEN - 1) {
            const float inv = __builtin_amdgcn_rcpf(S);
            // B fragments: pure in-lane packing. B[kt] halves h:
            // K = 32kt + 16*half + h = states of tiles jt=2kt (h0-7), 2kt+1.
            v16h B0, B1;
#pragma unroll
            for (int r = 0; r < 8; ++r) {
                B0[r]     = (_Float16)(q[0][r] * inv);
                B0[8 + r] = (_Float16)(q[1][r] * inv);
                B1[r]     = (_Float16)(q[2][r] * inv);
                B1[8 + r] = (_Float16)(q[3][r] * inv);
            }
#pragma unroll
            for (int jt = 0; jt < 4; ++jt) {
                v8f acc = {};
                acc = __builtin_amdgcn_wmma_f32_16x16x32_f16(
                        false, Af[jt][0], false, B0, (short)0, acc, false, false);
                acc = __builtin_amdgcn_wmma_f32_16x16x32_f16(
                        false, Af[jt][1], false, B1, (short)0, acc, false, false);
#pragma unroll
                for (int r = 0; r < 8; ++r) mid[jt][r] = acc[r];  // stay linear
            }
#pragma unroll
            for (int i = 0; i < 32; ++i) Ecur[i] = Enext[i];
        }
    }

    // Each lane accumulated its own batch column (duplicated across halves).
    const float tot = halfrow_sum(pacc);
    if (lane == 0) atomicAdd(out, tot);
}

// ---------------------------------------------------------------------------
extern "C" void kernel_launch(void* const* d_in, const int* in_sizes, int n_in,
                              void* d_out, int out_size, void* d_ws, size_t ws_size,
                              hipStream_t stream) {
    const int*   sent  = (const int*)  d_in[0];
    const float* masks = (const float*)d_in[1];
    const float* emb   = (const float*)d_in[2];
    const float* trans = (const float*)d_in[3];
    const float* begin = (const float*)d_in[4];
    float* out = (float*)d_out;
    float* wsf = (float*)d_ws;

    k_logz_part <<<125, 256, 0, stream>>>(emb, wsf + WS_PM, wsf + WS_PS);
    k_logz_final<<<1,   64,  0, stream>>>(wsf + WS_PM, wsf + WS_PS, wsf + WS_LOGZ);
    k_prep      <<<1,   64,  0, stream>>>(trans, begin, wsf, out);

    const size_t need = (size_t)(WS_EXP + (size_t)VOCAB * NS) * sizeof(float);
    if (ws_size >= need) {
        k_expemb<<<(VOCAB * NS) / 256, 256, 0, stream>>>(emb, wsf + WS_LOGZ,
                                                         wsf + WS_EXP);
        k_hmm<true> <<<16, 32, 0, stream>>>(sent, masks, emb, wsf, out);
    } else {
        k_hmm<false><<<16, 32, 0, stream>>>(sent, masks, emb, wsf, out);
    }
}
